// StateSpaceMixer_80204219285710
// MI455X (gfx1250) — matrix operand order repaired and checked by reading, not compiled, not run on hardware
//
#include <hip/hip_runtime.h>
#include <hip/hip_bf16.h>

// ---------------------------------------------------------------------------
// StateSpaceMixer fused bf16-WMMA implementation for gfx1250 (MI455X).
//   DIM=1024, STATE=2048, T=4096, B=4, K(conv)=4
// Workspace layout (bytes):
//   [0,        8388608)  w_inT  bf16 [4096][1024]   (w_in transposed)
//   [8388608, 12582912)  w_outT bf16 [1024][2048]   (w_out transposed)
//   [12582912,79691776)  mixed  bf16 [16384][2048]
// ---------------------------------------------------------------------------

typedef __attribute__((ext_vector_type(16))) __bf16 v16bf;
typedef __attribute__((ext_vector_type(8)))  float  v8f;

union Frag16 { v16bf v; uint4 q[2]; };

__device__ __forceinline__ unsigned short f2bf(float f) {
    union { float f; unsigned u; } v; v.f = f;
    unsigned r = v.u + 0x7FFFu + ((v.u >> 16) & 1u);   // round-to-nearest-even
    return (unsigned short)(r >> 16);
}

// ---------------------------------------------------------------------------
// Transpose + downconvert: dst[c][r] = bf16(src[r][c]); src is R x C row-major.
// ---------------------------------------------------------------------------
__global__ __launch_bounds__(256) void tposeK(const float* __restrict__ src,
                                              unsigned short* __restrict__ dst,
                                              int R, int C) {
    __shared__ float tile[32][33];
    int c0 = blockIdx.x * 32, r0 = blockIdx.y * 32;
    int tx = threadIdx.x & 31, ty = threadIdx.x >> 5;      // 32 x 8
#pragma unroll
    for (int i = 0; i < 32; i += 8) {
        int r = r0 + ty + i, c = c0 + tx;
        tile[ty + i][tx] = (r < R && c < C) ? src[(size_t)r * C + c] : 0.f;
    }
    __syncthreads();
#pragma unroll
    for (int i = 0; i < 32; i += 8) {
        int cc = c0 + ty + i;          // dst row
        int rr = r0 + tx;              // dst col
        if (cc < C && rr < R) dst[(size_t)cc * R + rr] = f2bf(tile[tx][ty + i]);
    }
}

// ---------------------------------------------------------------------------
// K1: fused  proj = x@w_in (+b_in) -> causal depthwise conv -> sig*tanh -> mixed (bf16)
// One WG = 64 time rows of one batch (+3 halo rows recomputed).
// All 8 waves compute the SAME 5 compile-time M-subtiles so accumulators stay
// in VGPRs; per k-step, all 12 ds_loads are issued before the 5 WMMAs so the
// compiler can overlap LDS latency with matrix ops (partial dscnt waits).
// LDS:
//   sA  : 80 rows x 1032 bf16 (stride-padded)            = 165120 B
//   region R (reused):
//     B piece:  sBg[64][136] + sBs[64][136] bf16         =  34816 B
//     staging:  sG[80][68] f32 + sS[80][68] f32          =  43520 B
//   total smem = 165120 + 43520 = 208640 B
// ---------------------------------------------------------------------------
#define ASTR   1032
#define A_BYTES (80 * ASTR * 2)
#define BSTR   136
#define CSTR   68
#define K1_SMEM (A_BYTES + 2 * 80 * CSTR * 4)

__global__ __launch_bounds__(256) void mixerK1(const float* __restrict__ x,
                                               const unsigned short* __restrict__ w_inT,
                                               const float* __restrict__ b_in,
                                               const float* __restrict__ w_conv,
                                               const float* __restrict__ b_conv,
                                               unsigned short* __restrict__ mixed) {
    extern __shared__ char smem[];
    unsigned short* sA  = (unsigned short*)smem;
    unsigned short* sBg = (unsigned short*)(smem + A_BYTES);
    unsigned short* sBs = sBg + 64 * BSTR;
    float* sG = (float*)(smem + A_BYTES);
    float* sS = sG + 80 * CSTR;

    const int tid  = threadIdx.x;
    const int wave = tid >> 5;
    const int lane = tid & 31;
    const int tile = blockIdx.x;
    const int bidx = tile >> 6;                 // 64 tiles per batch
    const int t0   = (tile & 63) * 64;
    const float* xb = x + (size_t)bidx * 4096 * 1024;

    // ---- stage x tile (rows t0-3 .. t0+63 -> LDS rows 0..66; 67..79 zero) ----
    for (int q = tid; q < 80 * 256; q += 256) {
        int i  = q >> 8;
        int c4 = (q & 255) * 4;
        int t  = t0 - 3 + i;
        float4 v = make_float4(0.f, 0.f, 0.f, 0.f);
        if (i < 67 && t >= 0 && t < 4096)
            v = *(const float4*)(xb + (size_t)t * 1024 + c4);
        uint2 pk;
        pk.x = (unsigned)f2bf(v.x) | ((unsigned)f2bf(v.y) << 16);
        pk.y = (unsigned)f2bf(v.z) | ((unsigned)f2bf(v.w) << 16);
        *(uint2*)(sA + (size_t)i * ASTR + c4) = pk;
    }
    __syncthreads();

    const bool isState = (wave >= 4);
    const int  nsub    = wave & 3;              // 16-col subtile within 64-col chunk

    const v8f zacc = {0.f,0.f,0.f,0.f,0.f,0.f,0.f,0.f};

    for (int j = 0; j < 32; ++j) {              // 64-col chunk pairs
        v8f acc[5];
#pragma unroll
        for (int m = 0; m < 5; ++m) acc[m] = zacc;
        const int nG = j * 64;                  // gate cols in proj
        const int nS = 2048 + j * 64;           // state cols in proj

        for (int kp = 0; kp < 1024; kp += 128) {
            __syncthreads();                    // region R free (prev piece / prev conv done)
            {   // load B piece: 128 rows (64 gate + 64 state), 2 threads/row
                int row = tid >> 1, half = tid & 1;
                int srow = (row < 64) ? (nG + row) : (nS + row - 64);
                const uint4* src = (const uint4*)(w_inT + (size_t)srow * 1024 + kp + half * 64);
                unsigned short* db = (row < 64) ? (sBg + row * BSTR + half * 64)
                                                : (sBs + (row - 64) * BSTR + half * 64);
                uint4* d = (uint4*)db;
#pragma unroll
                for (int q = 0; q < 8; ++q) d[q] = src[q];
                if (kp + 128 < 1024)            // global_prefetch_b8 of next piece
                    __builtin_prefetch((const void*)(w_inT + (size_t)srow * 1024 + kp + 128 + half * 64), 0, 1);
            }
            __syncthreads();

#pragma unroll
            for (int s = 0; s < 4; ++s) {       // 4 k-steps of 32
                Frag16 bfr, af[5];
                {   // B fragment: lane n=l&15, k=kloc + (l>>4)*16 .. +15  (contiguous k order, same as A)
                    int n    = nsub * 16 + (lane & 15);
                    int kloc = s * 32 + ((lane >> 4) * 16);
                    const unsigned short* p = (isState ? sBs : sBg) + n * BSTR + kloc;
                    bfr.q[0] = *(const uint4*)p;
                    bfr.q[1] = *(const uint4*)(p + 8);
                }
                int kk = kp + s * 32;
                // issue all A-fragment loads first (overlap with WMMAs below)
#pragma unroll
                for (int m = 0; m < 5; ++m) {   // A frag: row=m*16+(l&15), k = kk+(l>>4)*16 .. +15
                    int row = m * 16 + (lane & 15);
                    int kb  = kk + ((lane >> 4) * 16);
                    const unsigned short* p = sA + row * ASTR + kb;
                    af[m].q[0] = *(const uint4*)p;
                    af[m].q[1] = *(const uint4*)(p + 8);
                }
#pragma unroll
                for (int m = 0; m < 5; ++m)
                    acc[m] = __builtin_amdgcn_wmma_f32_16x16x32_bf16(
                        false, af[m].v, false, bfr.v, (short)0, acc[m], false, false);
            }
        }
        __syncthreads();                        // done reading B region

        // ---- stage D fragments (f32) into region R ----
        {
            float* dst = isState ? sS : sG;
#pragma unroll
            for (int m = 0; m < 5; ++m)
#pragma unroll
                for (int r = 0; r < 8; ++r) {
                    int row = m * 16 + r + ((lane >> 4) * 8);   // 0..79; t = t0-3+row
                    int col = nsub * 16 + (lane & 15);
                    dst[row * CSTR + col] = acc[m][r];
                }
        }
        __syncthreads();

        // ---- causal conv(K=4) + sigmoid*tanh -> mixed chunk ----
        {
            int c  = tid & 63;
            int rg = tid >> 6;                  // 0..3
            int cG = j * 64 + c;                // state channel / gate col
            float big = b_in[cG];
            float bis = b_in[2048 + cG];
            float bcv = b_conv[cG];
            float w0 = w_conv[cG * 4 + 0], w1 = w_conv[cG * 4 + 1];
            float w2 = w_conv[cG * 4 + 2], w3 = w_conv[cG * 4 + 3];
#pragma unroll
            for (int rr = 0; rr < 16; ++rr) {
                int tt = rg * 16 + rr;
                float g  = sG[(tt + 3) * CSTR + c] + big;       // staging row i = tt+3
                float sv = bcv;
                int tb = t0 + tt - 3;           // global t of first tap
                if (tb + 0 >= 0) sv += w0 * (sS[(tt + 0) * CSTR + c] + bis);
                if (tb + 1 >= 0) sv += w1 * (sS[(tt + 1) * CSTR + c] + bis);
                if (tb + 2 >= 0) sv += w2 * (sS[(tt + 2) * CSTR + c] + bis);
                sv += w3 * (sS[(tt + 3) * CSTR + c] + bis);     // t itself, always valid
                float mval = (1.f / (1.f + __expf(-g))) * tanhf(sv);
                mixed[(size_t)(bidx * 4096 + t0 + tt) * 2048 + cG] = f2bf(mval);
            }
        }
        // leading __syncthreads of next kp loop protects region R
    }
}

// ---------------------------------------------------------------------------
// K2: out[16384][1024] = mixed[16384][2048] @ w_out + b_out (B from w_outT)
// WG tile 128x128, 8 waves (2x4), K piece = 128.  smem = 2 * 128*136*2 = 69632 B
// ---------------------------------------------------------------------------
#define K2_SMEM (2 * 128 * BSTR * 2)

__global__ __launch_bounds__(256) void mixerK2(const unsigned short* __restrict__ mixed,
                                               const unsigned short* __restrict__ w_outT,
                                               const float* __restrict__ b_out,
                                               float* __restrict__ out) {
    extern __shared__ char smem[];
    unsigned short* sA = (unsigned short*)smem;
    unsigned short* sB = sA + 128 * BSTR;

    const int tid  = threadIdx.x;
    const int wave = tid >> 5;
    const int lane = tid & 31;
    const int m0 = blockIdx.x * 128;
    const int n0 = blockIdx.y * 128;
    const int wm = wave >> 2;                   // 0..1 -> 64 rows each
    const int wn = wave & 3;                    // 0..3 -> 32 cols each

    const v8f zacc = {0.f,0.f,0.f,0.f,0.f,0.f,0.f,0.f};
    v8f acc[4][2];
#pragma unroll
    for (int mm = 0; mm < 4; ++mm)
#pragma unroll
        for (int nn = 0; nn < 2; ++nn) acc[mm][nn] = zacc;

    for (int kp = 0; kp < 2048; kp += 128) {
        __syncthreads();
        {   // stage A and B pieces: 2 threads per row, 128B halves
            int row = tid >> 1, half = tid & 1;
            const uint4* srcA = (const uint4*)(mixed  + (size_t)(m0 + row) * 2048 + kp + half * 64);
            const uint4* srcB = (const uint4*)(w_outT + (size_t)(n0 + row) * 2048 + kp + half * 64);
            uint4* dA = (uint4*)(sA + row * BSTR + half * 64);
            uint4* dB = (uint4*)(sB + row * BSTR + half * 64);
#pragma unroll
            for (int q = 0; q < 8; ++q) { dA[q] = srcA[q]; dB[q] = srcB[q]; }
            if (kp + 128 < 2048) {              // global_prefetch_b8 of next piece
                __builtin_prefetch((const void*)(mixed  + (size_t)(m0 + row) * 2048 + kp + 128 + half * 64), 0, 1);
                __builtin_prefetch((const void*)(w_outT + (size_t)(n0 + row) * 2048 + kp + 128 + half * 64), 0, 1);
            }
        }
        __syncthreads();

#pragma unroll
        for (int s = 0; s < 4; ++s) {
            Frag16 bfr[2], af[4];
#pragma unroll
            for (int nn = 0; nn < 2; ++nn) {
                int n    = wn * 32 + nn * 16 + (lane & 15);
                int kloc = s * 32 + ((lane >> 4) * 16);
                const unsigned short* p = sB + n * BSTR + kloc;
                bfr[nn].q[0] = *(const uint4*)p;
                bfr[nn].q[1] = *(const uint4*)(p + 8);
            }
            // issue all A-fragment loads before the WMMA burst
#pragma unroll
            for (int mm = 0; mm < 4; ++mm) {
                int row = wm * 64 + mm * 16 + (lane & 15);
                int kb  = s * 32 + ((lane >> 4) * 16);
                const unsigned short* p = sA + row * BSTR + kb;
                af[mm].q[0] = *(const uint4*)p;
                af[mm].q[1] = *(const uint4*)(p + 8);
            }
#pragma unroll
            for (int mm = 0; mm < 4; ++mm)
#pragma unroll
                for (int nn = 0; nn < 2; ++nn)
                    acc[mm][nn] = __builtin_amdgcn_wmma_f32_16x16x32_bf16(
                        false, af[mm].v, false, bfr[nn].v, (short)0, acc[mm][nn], false, false);
        }
    }

    // epilogue: + b_out, store f32
#pragma unroll
    for (int mm = 0; mm < 4; ++mm)
#pragma unroll
        for (int nn = 0; nn < 2; ++nn) {
            int col = n0 + wn * 32 + nn * 16 + (lane & 15);
            float bo = b_out[col];
#pragma unroll
            for (int r = 0; r < 8; ++r) {
                int row = m0 + wm * 64 + mm * 16 + r + ((lane >> 4) * 8);
                out[(size_t)row * 1024 + col] = acc[mm][nn][r] + bo;
            }
        }
}

// ---------------------------------------------------------------------------
extern "C" void kernel_launch(void* const* d_in, const int* in_sizes, int n_in,
                              void* d_out, int out_size, void* d_ws, size_t ws_size,
                              hipStream_t stream) {
    const float* x      = (const float*)d_in[0];
    const float* w_in   = (const float*)d_in[1];
    const float* b_in   = (const float*)d_in[2];
    const float* w_conv = (const float*)d_in[3];
    const float* b_conv = (const float*)d_in[4];
    const float* w_out  = (const float*)d_in[5];
    const float* b_out  = (const float*)d_in[6];
    float* out = (float*)d_out;

    char* ws = (char*)d_ws;
    unsigned short* w_inT  = (unsigned short*)(ws);
    unsigned short* w_outT = (unsigned short*)(ws + 8388608);
    unsigned short* mixed  = (unsigned short*)(ws + 12582912);

    // bf16 transposed weight copies
    tposeK<<<dim3(4096 / 32, 1024 / 32), 256, 0, stream>>>(w_in,  w_inT,  1024, 4096);
    tposeK<<<dim3(1024 / 32, 2048 / 32), 256, 0, stream>>>(w_out, w_outT, 2048, 1024);

    // fused GEMM1 + conv + activation
    mixerK1<<<dim3(256), 256, K1_SMEM, stream>>>(x, w_inT, b_in, w_conv, b_conv, mixed);

    // output GEMM
    mixerK2<<<dim3(128, 8), 256, K2_SMEM, stream>>>(mixed, w_outT, b_out, out);
}
